// DisentangledRelationalCrossAttention_51230369907056
// MI455X (gfx1250) — matrix operand order repaired and checked by reading, not compiled, not run on hardware
//
#include <hip/hip_runtime.h>

#define LSEQ   2048
#define DMODEL 1024
#define NHEADS 16
#define HDIM   64
#define QK_SCALE 0.125f   // 1/sqrt(64)

typedef __attribute__((ext_vector_type(16))) __bf16 v16bf;
typedef __attribute__((ext_vector_type(8)))  __bf16 v8bf;
typedef __attribute__((ext_vector_type(8)))  float  v8f;

// ---------------- device helpers ----------------

__device__ __forceinline__ v8f vzero8() {
  v8f z;
#pragma unroll
  for (int i = 0; i < 8; ++i) z[i] = 0.f;
  return z;
}

// A-fragment (16x32 bf16, row-major source), contiguous K order:
// lane<16 : row=lane,    elems 0..15 = K k..k+15
// lane>=16: row=lane-16, elems 0..15 = K k+16..k+31
// Caller passes p already offset by (row*ld + (hi?16:0)).
__device__ __forceinline__ v16bf frag_a_load(const __bf16* p) {
  return *(const v16bf*)p;
}

__device__ __forceinline__ v8f wmma_bf16(v16bf a, v16bf b, v8f c) {
  return __builtin_amdgcn_wmma_f32_16x16x32_bf16(
      /*neg_a=*/false, a, /*neg_b=*/false, b,
      /*c_mod=*/(short)0, c, /*reuse_a=*/false, /*reuse_b=*/false);
}

// width-16 reductions (rows of a 16x16 accumulator tile live across 16 lanes)
__device__ __forceinline__ float rmax16(float x) {
#pragma unroll
  for (int m = 1; m < 16; m <<= 1) x = fmaxf(x, __shfl_xor(x, m, 16));
  return x;
}
__device__ __forceinline__ float rsum16(float x) {
#pragma unroll
  for (int m = 1; m < 16; m <<= 1) x += __shfl_xor(x, m, 16);
  return x;
}

// ---------------- elementwise kernels ----------------

__global__ __launch_bounds__(256) void f32_to_bf16_kernel(const float* __restrict__ s,
                                                          __bf16* __restrict__ d, int n) {
  int i = blockIdx.x * 256 + threadIdx.x;
  if (i < n) d[i] = (__bf16)s[i];
}

// Tiled transpose: WT[n*K + k] = (bf16) W[k*N + n]. K, N multiples of 32.
// Coalesced reads and writes via a 32x33 LDS tile.
__global__ __launch_bounds__(256) void transpose_to_bf16_kernel(const float* __restrict__ W,
                                                                __bf16* __restrict__ WT,
                                                                int K, int N) {
  __shared__ __bf16 t[32][33];
  const int n0 = blockIdx.x << 5;
  const int k0 = blockIdx.y << 5;
  const int tx = threadIdx.x & 31;
  const int ty = threadIdx.x >> 5;     // 0..7
#pragma unroll
  for (int r = 0; r < 32; r += 8)
    t[ty + r][tx] = (__bf16)W[(size_t)(k0 + ty + r) * N + n0 + tx];
  __syncthreads();
#pragma unroll
  for (int r = 0; r < 32; r += 8)
    WT[(size_t)(n0 + ty + r) * K + k0 + tx] = t[tx][ty + r];
}

// RoPE on (L, nheads, 64) fp32 -> bf16 (interleaved real/imag pairs)
__global__ __launch_bounds__(256) void rope_bf16_kernel(const float* __restrict__ src,
                                                        __bf16* __restrict__ dst,
                                                        const float* __restrict__ fcos,
                                                        const float* __restrict__ fsin,
                                                        int nheads) {
  int idx = blockIdx.x * 256 + threadIdx.x;    // L * nheads * 32
  if (idx >= LSEQ * nheads * 32) return;
  int t   = idx & 31;
  int tmp = idx >> 5;
  int hh  = tmp % nheads;
  int pos = tmp / nheads;
  float c = fcos[pos * 32 + t];
  float s = fsin[pos * 32 + t];
  size_t base = ((size_t)pos * nheads + hh) * HDIM + 2 * t;
  float xr = src[base], xi = src[base + 1];
  dst[base]     = (__bf16)(xr * c - xi * s);
  dst[base + 1] = (__bf16)(xr * s + xi * c);
}

// ---------------- bf16 WMMA GEMM ----------------
// C(MxN,f32) = A(MxK,bf16) * B(KxN), B given transposed as BT(NxK).
// One wave computes a 16x64 tile: one A fragment reused across 4 B fragments.
// Requires N % 64 == 0, K % 32 == 0.
__global__ __launch_bounds__(32) void gemm_bf16_kernel(const __bf16* __restrict__ A,
                                                       const __bf16* __restrict__ BT,
                                                       float* __restrict__ C,
                                                       int M, int N, int K) {
  const int lane = threadIdx.x;
  const int lr = lane & 15;
  const int hi = lane >> 4;
  const int ntn = N >> 6;                 // N / 64
  const int tile = blockIdx.x;
  const int m0 = (tile / ntn) << 4;
  const int n0 = (tile % ntn) << 6;
  if (m0 >= M) return;

  const __bf16* pa = A + (size_t)(m0 + lr) * K + (hi ? 16 : 0);
  const __bf16* pb[4];
#pragma unroll
  for (int nt = 0; nt < 4; ++nt)
    pb[nt] = BT + (size_t)(n0 + nt * 16 + lr) * K + (hi ? 16 : 0);

  v8f acc[4];
#pragma unroll
  for (int nt = 0; nt < 4; ++nt) acc[nt] = vzero8();

  for (int k = 0; k < K; k += 32) {
    __builtin_prefetch((const void*)(pa + k + 512), 0, 3);   // near (WGP) prefetch
    v16bf a = frag_a_load(pa + k);
#pragma unroll
    for (int nt = 0; nt < 4; ++nt) {
      __builtin_prefetch((const void*)(pb[nt] + k + 512), 0, 3);
      acc[nt] = wmma_bf16(a, *(const v16bf*)(pb[nt] + k), acc[nt]);
    }
  }

#pragma unroll
  for (int nt = 0; nt < 4; ++nt) {
    float* pc = C + (size_t)(m0 + 8 * hi) * N + n0 + nt * 16 + lr;
#pragma unroll
    for (int v = 0; v < 8; ++v) pc[(size_t)v * N] = acc[nt][v];
  }
}

// ---------------- fused relational attention ----------------
// One wave per (head, 16-query tile). Writes rel_scores (all j), attn_scores
// (softmax, zeros past diagonal), and per-head output (2048 x 1024) fp32.
__global__ __launch_bounds__(32) void rca_attention_kernel(
    const __bf16* __restrict__ qa, const __bf16* __restrict__ ka,
    const __bf16* __restrict__ qr, const __bf16* __restrict__ kr,
    const __bf16* __restrict__ svT,                 // (64 x 2048) bf16
    float* __restrict__ attn_out, float* __restrict__ rel_out,
    float* __restrict__ oh) {
  __shared__ __attribute__((aligned(32))) __bf16 prtile[16 * 32];

  const int lane = threadIdx.x;
  const int lr = lane & 15;
  const int hi = lane >> 4;
  const int h  = blockIdx.x >> 7;    // / 128
  const int it = blockIdx.x & 127;
  const int i0 = it << 4;
  const float NEG_INF = -__builtin_huge_valf();

  // Q fragments (K = 64 -> two 32-chunks each)
  const __bf16* pqa = qa + (size_t)(i0 + lr) * DMODEL + h * HDIM + (hi ? 16 : 0);
  const __bf16* pqr = qr + (size_t)(i0 + lr) * DMODEL + h * HDIM + (hi ? 16 : 0);
  const v16bf aq0 = frag_a_load(pqa);
  const v16bf aq1 = frag_a_load(pqa + 32);
  const v16bf ar0 = frag_a_load(pqr);
  const v16bf ar1 = frag_a_load(pqr + 32);

  float mrow[8], lrow[8];
#pragma unroll
  for (int v = 0; v < 8; ++v) { mrow[v] = NEG_INF; lrow[v] = 0.f; }

  const size_t srow0 = ((size_t)h * LSEQ + i0) * LSEQ;

  // ---- pass 1: rel scores (all j) + online softmax stats (j <= i) ----
  for (int jt = 0; jt < LSEQ / 16; ++jt) {
    const int j0 = jt << 4;
    const __bf16* pkr = kr + (size_t)(j0 + lr) * HDIM + (hi ? 16 : 0);
    v8f accr = vzero8();
    accr = wmma_bf16(ar0, *(const v16bf*)pkr, accr);
    accr = wmma_bf16(ar1, *(const v16bf*)(pkr + 32), accr);
#pragma unroll
    for (int v = 0; v < 8; ++v)
      rel_out[srow0 + (size_t)(v + 8 * hi) * LSEQ + j0 + lr] = accr[v] * QK_SCALE;

    if (jt <= it) {
      const __bf16* pka = ka + (size_t)(j0 + lr) * HDIM + (hi ? 16 : 0);
      v8f acca = vzero8();
      acca = wmma_bf16(aq0, *(const v16bf*)pka, acca);
      acca = wmma_bf16(aq1, *(const v16bf*)(pka + 32), acca);
      const int j = j0 + lr;
#pragma unroll
      for (int v = 0; v < 8; ++v) {
        const int i = i0 + v + 8 * hi;
        float s = (j <= i) ? acca[v] * QK_SCALE : NEG_INF;
        float mnew = fmaxf(mrow[v], rmax16(s));
        float e = __expf(s - mnew);
        float esum = rsum16(e);
        lrow[v] = lrow[v] * __expf(mrow[v] - mnew) + esum;
        mrow[v] = mnew;
      }
    }
  }

  float linv[8];
#pragma unroll
  for (int v = 0; v < 8; ++v) linv[v] = 1.f / lrow[v];

  // ---- pass 2: write normalized attn, accumulate (attn*rel) @ sv ----
  v8f accO[4];
#pragma unroll
  for (int t = 0; t < 4; ++t) accO[t] = vzero8();

  for (int jc = 0; jc < LSEQ; jc += 32) {
    const int jt0 = jc >> 4;
    if (jt0 > it) {
#pragma unroll
      for (int sub = 0; sub < 2; ++sub)
#pragma unroll
        for (int v = 0; v < 8; ++v)
          attn_out[srow0 + (size_t)(v + 8 * hi) * LSEQ + jc + sub * 16 + lr] = 0.f;
      continue;
    }
#pragma unroll
    for (int sub = 0; sub < 2; ++sub) {
      const int jt = jt0 + sub;
      const int j0 = jt << 4;
      if (jt <= it) {
        const __bf16* pka = ka + (size_t)(j0 + lr) * HDIM + (hi ? 16 : 0);
        const __bf16* pkr = kr + (size_t)(j0 + lr) * HDIM + (hi ? 16 : 0);
        v8f acca = vzero8(), accr = vzero8();
        acca = wmma_bf16(aq0, *(const v16bf*)pka, acca);
        acca = wmma_bf16(aq1, *(const v16bf*)(pka + 32), acca);
        accr = wmma_bf16(ar0, *(const v16bf*)pkr, accr);
        accr = wmma_bf16(ar1, *(const v16bf*)(pkr + 32), accr);
        const int j = j0 + lr;
#pragma unroll
        for (int v = 0; v < 8; ++v) {
          const int i = i0 + v + 8 * hi;
          float s = (j <= i) ? acca[v] * QK_SCALE : NEG_INF;
          float p = __expf(s - mrow[v]) * linv[v];   // 0 for masked
          attn_out[srow0 + (size_t)(v + 8 * hi) * LSEQ + j0 + lr] = p;
          prtile[(v + 8 * hi) * 32 + sub * 16 + lr] = (__bf16)(p * (accr[v] * QK_SCALE));
        }
      } else {
#pragma unroll
        for (int v = 0; v < 8; ++v) {
          attn_out[srow0 + (size_t)(v + 8 * hi) * LSEQ + j0 + lr] = 0.f;
          prtile[(v + 8 * hi) * 32 + sub * 16 + lr] = (__bf16)0.f;
        }
      }
    }
    __syncthreads();
    const v16bf apr = frag_a_load(&prtile[lr * 32 + (hi ? 16 : 0)]);
    __syncthreads();
#pragma unroll
    for (int dt = 0; dt < 4; ++dt) {
      const __bf16* pb = svT + (size_t)(dt * 16 + lr) * LSEQ + jc + (hi ? 16 : 0);
      accO[dt] = wmma_bf16(apr, *(const v16bf*)pb, accO[dt]);
    }
  }

#pragma unroll
  for (int dt = 0; dt < 4; ++dt)
#pragma unroll
    for (int v = 0; v < 8; ++v)
      oh[(size_t)(i0 + v + 8 * hi) * DMODEL + h * HDIM + dt * 16 + lr] = accO[dt][v];
}

// ---------------- host launcher ----------------

extern "C" void kernel_launch(void* const* d_in, const int* in_sizes, int n_in,
                              void* d_out, int out_size, void* d_ws, size_t ws_size,
                              hipStream_t stream) {
  (void)in_sizes; (void)n_in; (void)out_size; (void)ws_size;
  const float* x    = (const float*)d_in[0];
  const float* sym  = (const float*)d_in[1];
  const float* fcos = (const float*)d_in[2];
  const float* fsin = (const float*)d_in[3];
  const float* wqa  = (const float*)d_in[4];
  const float* wka  = (const float*)d_in[5];
  const float* wqr  = (const float*)d_in[6];
  const float* wkr  = (const float*)d_in[7];
  const float* wv   = (const float*)d_in[8];
  const float* wo   = (const float*)d_in[9];

  float* out      = (float*)d_out;                                   // (2048,1024)
  float* attn_out = out + (size_t)LSEQ * DMODEL;                      // (16,2048,2048)
  float* rel_out  = attn_out + (size_t)NHEADS * LSEQ * LSEQ;          // (16,2048,2048)

  size_t off = 0;
  auto alloc = [&](size_t bytes) {
    void* p = (char*)d_ws + off;
    off += (bytes + 255) & ~(size_t)255;
    return p;
  };
  __bf16* xb   = (__bf16*)alloc((size_t)LSEQ * DMODEL * 2);
  __bf16* sb   = (__bf16*)alloc((size_t)LSEQ * DMODEL * 2);
  __bf16* wqaT = (__bf16*)alloc((size_t)DMODEL * DMODEL * 2);
  __bf16* wkaT = (__bf16*)alloc((size_t)DMODEL * HDIM * 2);
  __bf16* wqrT = (__bf16*)alloc((size_t)DMODEL * DMODEL * 2);
  __bf16* wkrT = (__bf16*)alloc((size_t)DMODEL * HDIM * 2);
  __bf16* wvT  = (__bf16*)alloc((size_t)DMODEL * HDIM * 2);
  __bf16* woT  = (__bf16*)alloc((size_t)DMODEL * DMODEL * 2);
  float* qa_f  = (float*)alloc((size_t)LSEQ * DMODEL * 4);
  float* ka_f  = (float*)alloc((size_t)LSEQ * HDIM * 4);
  float* qr_f  = (float*)alloc((size_t)LSEQ * DMODEL * 4);
  float* kr_f  = (float*)alloc((size_t)LSEQ * HDIM * 4);
  float* sv_f  = (float*)alloc((size_t)LSEQ * HDIM * 4);
  __bf16* qa_b = (__bf16*)alloc((size_t)LSEQ * DMODEL * 2);
  __bf16* ka_b = (__bf16*)alloc((size_t)LSEQ * HDIM * 2);
  __bf16* qr_b = (__bf16*)alloc((size_t)LSEQ * DMODEL * 2);
  __bf16* kr_b = (__bf16*)alloc((size_t)LSEQ * HDIM * 2);
  __bf16* svT_b = (__bf16*)alloc((size_t)HDIM * LSEQ * 2);
  float* oh_f  = (float*)alloc((size_t)LSEQ * DMODEL * 4);
  __bf16* oh_b = (__bf16*)alloc((size_t)LSEQ * DMODEL * 2);

  const int nXD = LSEQ * DMODEL;          // 2,097,152
  const int nLH = LSEQ * HDIM;            // 131,072

  // --- prep: converts + tiled weight transposes ---
  f32_to_bf16_kernel<<<(nXD + 255) / 256, 256, 0, stream>>>(x, xb, nXD);
  f32_to_bf16_kernel<<<(nXD + 255) / 256, 256, 0, stream>>>(sym, sb, nXD);
  transpose_to_bf16_kernel<<<dim3(DMODEL / 32, DMODEL / 32), 256, 0, stream>>>(wqa, wqaT, DMODEL, DMODEL);
  transpose_to_bf16_kernel<<<dim3(HDIM / 32, DMODEL / 32), 256, 0, stream>>>(wka, wkaT, DMODEL, HDIM);
  transpose_to_bf16_kernel<<<dim3(DMODEL / 32, DMODEL / 32), 256, 0, stream>>>(wqr, wqrT, DMODEL, DMODEL);
  transpose_to_bf16_kernel<<<dim3(HDIM / 32, DMODEL / 32), 256, 0, stream>>>(wkr, wkrT, DMODEL, HDIM);
  transpose_to_bf16_kernel<<<dim3(HDIM / 32, DMODEL / 32), 256, 0, stream>>>(wv, wvT, DMODEL, HDIM);
  transpose_to_bf16_kernel<<<dim3(DMODEL / 32, DMODEL / 32), 256, 0, stream>>>(wo, woT, DMODEL, DMODEL);

  // --- projections (WMMA GEMMs, 16x64 tile per wave) ---
  gemm_bf16_kernel<<<(LSEQ / 16) * (DMODEL / 64), 32, 0, stream>>>(xb, wqaT, qa_f, LSEQ, DMODEL, DMODEL);
  gemm_bf16_kernel<<<(LSEQ / 16) * (HDIM / 64), 32, 0, stream>>>(xb, wkaT, ka_f, LSEQ, HDIM, DMODEL);
  gemm_bf16_kernel<<<(LSEQ / 16) * (DMODEL / 64), 32, 0, stream>>>(xb, wqrT, qr_f, LSEQ, DMODEL, DMODEL);
  gemm_bf16_kernel<<<(LSEQ / 16) * (HDIM / 64), 32, 0, stream>>>(xb, wkrT, kr_f, LSEQ, HDIM, DMODEL);
  gemm_bf16_kernel<<<(LSEQ / 16) * (HDIM / 64), 32, 0, stream>>>(sb, wvT, sv_f, LSEQ, HDIM, DMODEL);

  // --- RoPE (q_a: 16 heads, k_a: 1 head) + straight converts (q_r, k_r) ---
  rope_bf16_kernel<<<(LSEQ * NHEADS * 32 + 255) / 256, 256, 0, stream>>>(qa_f, qa_b, fcos, fsin, NHEADS);
  rope_bf16_kernel<<<(LSEQ * 32 + 255) / 256, 256, 0, stream>>>(ka_f, ka_b, fcos, fsin, 1);
  f32_to_bf16_kernel<<<(nXD + 255) / 256, 256, 0, stream>>>(qr_f, qr_b, nXD);
  f32_to_bf16_kernel<<<(nLH + 255) / 256, 256, 0, stream>>>(kr_f, kr_b, nLH);
  // svT (64 x 2048): treat sv_f as (K=2048 rows, N=64 cols)
  transpose_to_bf16_kernel<<<dim3(HDIM / 32, LSEQ / 32), 256, 0, stream>>>(sv_f, svT_b, LSEQ, HDIM);

  // --- fused attention: rel scores, softmax attn, (attn*rel)@sv ---
  rca_attention_kernel<<<NHEADS * (LSEQ / 16), 32, 0, stream>>>(
      qa_b, ka_b, qr_b, kr_b, svT_b, attn_out, rel_out, oh_f);

  // --- output projection ---
  f32_to_bf16_kernel<<<(nXD + 255) / 256, 256, 0, stream>>>(oh_f, oh_b, nXD);
  gemm_bf16_kernel<<<(LSEQ / 16) * (DMODEL / 64), 32, 0, stream>>>(oh_b, woT, out, LSEQ, DMODEL, DMODEL);
}
